// Generator_60378650247596
// MI455X (gfx1250) — matrix operand order repaired and checked by reading, not compiled, not run on hardware
//
#include <hip/hip_runtime.h>
#include <hip/hip_bf16.h>

typedef __attribute__((ext_vector_type(16))) _Float16 v16h;
typedef __attribute__((ext_vector_type(8)))  _Float16 v8h;
typedef __attribute__((ext_vector_type(8)))  float    v8f;

#define BSZ   256
#define LSIG  8192
#define NZc   50
#define NTOKc 128
#define SEQc  258
#define NHEADc 5
#define DHEADc 10
#define NHIDc 512
#define MROWS (BSZ*SEQc)   /* 66048, multiple of 16 */

#define ARC_Sc  30.0f
#define COS_Mc  0.8775825618903728f
#define SIN_Mc  0.4794255386042030f
#define THc    -0.8775825618903728f
#define MMc     0.2397127693021015f
#define QK_SCALE 0.3162277660168379f  /* 1/sqrt(10) */

// ---------------- workspace layout (bytes) ----------------
#define REG_A  71303168ull     /* 68 MiB: enc ping A / FF hidden f16 / dec ping A */
#define REG_B  67108864ull     /* 64 MiB: enc ping B / q,k,v / dec ping B */
#define OFF_A   0ull
#define OFF_B   (OFF_A + REG_A)
#define OFF_X   (OFF_B + REG_B)                 /* x fp32 M x 50   : 13,209,600 B */
#define OFF_XH  (OFF_X  + 13209600ull)          /* xh f16 M x 64   :  8,454,144 B */
#define OFF_OH  (OFF_XH +  8454144ull)          /* attn out f16    :  8,454,144 B */
#define OFF_TMP (OFF_OH +  8454144ull)          /* proj tmp fp32   : 13,209,600 B */
#define OFF_WH  (OFF_TMP+ 13209600ull)          /* padded f16 wts  :    491,520 B */
#define OFF_Z   (OFF_WH +   491520ull)          /* arc z 2x256x50  :    102,400 B */
#define OFF_G1  (OFF_Z  +   102400ull)          /* gen1 fp32       :  8,388,608 B */
#define WH_PER_LAYER 81920   /* halves: WQ@0 WK@4096 WV@8192 WO@12288 W1@16384 W2@49152 */

static inline int GRD(size_t n) { return (int)((n + 255) / 256); }

// ================= encoder conv1d (k=4, s=2, p=1) =================
__global__ void k_conv1d(const float* __restrict__ in, const float* __restrict__ w,
                         float* __restrict__ out, int Cin, int Cout, int Lin, int leaky)
{
    int Lout = Lin >> 1;
    size_t idx = (size_t)blockIdx.x * blockDim.x + threadIdx.x;
    size_t total = (size_t)BSZ * Cout * Lout;
    if (idx >= total) return;
    int l  = (int)(idx % Lout);
    int co = (int)((idx / Lout) % Cout);
    int b  = (int)(idx / ((size_t)Lout * Cout));
    float acc = 0.f;
    int base = 2 * l - 1;
    for (int ci = 0; ci < Cin; ++ci) {
        const float* ip = in + ((size_t)b * Cin + ci) * Lin;
        const float* wp = w + ((size_t)co * Cin + ci) * 4;
#pragma unroll
        for (int j = 0; j < 4; ++j) {
            int t = base + j;
            if (t >= 0 && t < Lin) acc += ip[t] * wp[j];
        }
    }
    if (leaky) acc = acc > 0.f ? acc : 0.2f * acc;
    out[idx] = acc;
}

// ===== fp32 KxN -> padded TRANSPOSED f16 (Np x Kp, row-major over N) =====
// dst[n*Kp + k] = src[k*N + n]; zeros in the padding.  B-fragment loads become
// two contiguous 8-half (b128) reads in the GEMM.
__global__ void k_pad_f16t(const float* __restrict__ src, _Float16* __restrict__ dst,
                           int K, int N, int Kp, int Np)
{
    size_t idx = (size_t)blockIdx.x * blockDim.x + threadIdx.x;
    size_t total = (size_t)Kp * Np;
    if (idx >= total) return;
    int k = (int)(idx % Kp);
    int n = (int)(idx / Kp);
    dst[idx] = (k < K && n < N) ? (_Float16)src[(size_t)k * N + n] : (_Float16)0.f;
}

__global__ void k_zero_h(_Float16* p, size_t n)
{
    size_t idx = (size_t)blockIdx.x * blockDim.x + threadIdx.x;
    if (idx < n) p[idx] = (_Float16)0.f;
}

// ================= token assembly: cls + latents + pos =================
__global__ void k_build_x(const float* __restrict__ latS, const float* __restrict__ latF,
                          const float* __restrict__ clsS, const float* __restrict__ clsF,
                          const float* __restrict__ pos,
                          float* __restrict__ x, _Float16* __restrict__ xh)
{
    size_t idx = (size_t)blockIdx.x * blockDim.x + threadIdx.x;
    size_t total = (size_t)MROWS * 64;
    if (idx >= total) return;
    int c = (int)(idx % 64);
    size_t row = idx / 64;
    int s = (int)(row % SEQc);
    int b = (int)(row / SEQc);
    float v = 0.f;
    if (c < NZc) {
        float base;
        if (s == 0)            base = clsS[c];
        else if (s <= NTOKc)   base = latS[((size_t)b * NZc + c) * NTOKc + (s - 1)];
        else if (s == NTOKc+1) base = clsF[c];
        else                   base = latF[((size_t)b * NZc + c) * NTOKc + (s - NTOKc - 2)];
        v = base + pos[(size_t)s * NZc + c];
        x[row * NZc + c] = v;
    }
    xh[row * 64 + c] = (_Float16)v;
}

// ===== WMMA f16 GEMM: D = A(MxK) @ B(KxN) + bias; B given TRANSPOSED (NxK) =====
// One wave computes a 16x64 strip (4 N-tiles), reusing the A fragment across the
// 4 WMMAs per k-step.  All fragment loads are contiguous 128-bit.
__global__ void __launch_bounds__(128)
k_gemm_wmma(const _Float16* __restrict__ A, int lda,
            const _Float16* __restrict__ Bt, int ldbt,     /* row stride over N */
            const float* __restrict__ bias,
            float* __restrict__ outF, int ldoF,
            _Float16* __restrict__ outH, int ldoH,
            int M, int N, int K, int nValid, int relu)
{
    int wave = blockIdx.x * (blockDim.x >> 5) + (threadIdx.x >> 5);
    int groupsN = N >> 6;                          /* 4 tiles of 16 per wave */
    int groups  = (M >> 4) * groupsN;
    if (wave >= groups) return;                    // wave-uniform: EXEC stays all-1s
    int tileM = wave / groupsN;
    int n0    = (wave % groupsN) << 6;
    int lane = threadIdx.x & 31;
    int half = lane >> 4;                           // 0 or 1
    int l16  = lane & 15;

    v8f acc0 = {}, acc1 = {}, acc2 = {}, acc3 = {};
    const _Float16* arow = A + (size_t)(tileM * 16 + l16) * lda;
    const _Float16* brow = Bt + (size_t)(n0 + l16) * ldbt;

    for (int k0 = 0; k0 < K; k0 += 32) {
        // A 16x32 f16: lane<16 -> K {0..7,16..23}; lane>=16 -> K {8..15,24..31}
        const _Float16* ap = arow + k0 + half * 8;
        v8h a0 = *(const v8h*)ap;
        v8h a1 = *(const v8h*)(ap + 16);
        v16h a;
#pragma unroll
        for (int e = 0; e < 8; ++e) { a[e] = a0[e]; a[e + 8] = a1[e]; }
        // B 32x16 f16: lane<16 -> K {0..7,16..23} ; lane>=16 -> K {8..15,24..31} ; col = n0+t*16+(lane&15)
        // transposed layout -> two contiguous 8-half runs per lane per tile
#pragma unroll
        for (int t = 0; t < 4; ++t) {
            const _Float16* bp = brow + (size_t)t * 16 * ldbt + k0 + half * 8;
            v8h b0 = *(const v8h*)bp;
            v8h b1 = *(const v8h*)(bp + 16);
            v16h b;
#pragma unroll
            for (int e = 0; e < 8; ++e) { b[e] = b0[e]; b[e + 8] = b1[e]; }
            if (t == 0) acc0 = __builtin_amdgcn_wmma_f32_16x16x32_f16(false, a, false, b, (short)0, acc0, false, false);
            if (t == 1) acc1 = __builtin_amdgcn_wmma_f32_16x16x32_f16(false, a, false, b, (short)0, acc1, false, false);
            if (t == 2) acc2 = __builtin_amdgcn_wmma_f32_16x16x32_f16(false, a, false, b, (short)0, acc2, false, false);
            if (t == 3) acc3 = __builtin_amdgcn_wmma_f32_16x16x32_f16(false, a, false, b, (short)0, acc3, false, false);
        }
    }
#pragma unroll
    for (int t = 0; t < 4; ++t) {
        v8f acc = (t == 0) ? acc0 : (t == 1) ? acc1 : (t == 2) ? acc2 : acc3;
        int col = n0 + t * 16 + l16;
        if (col >= nValid) continue;
        float bs = bias ? bias[col] : 0.f;
#pragma unroll
        for (int r = 0; r < 8; ++r) {
            int row = tileM * 16 + r + half * 8;
            float v = acc[r] + bs;
            if (relu) v = v > 0.f ? v : 0.f;
            if (outF) outF[(size_t)row * ldoF + col] = v;
            if (outH) outH[(size_t)row * ldoH + col] = (_Float16)v;
        }
    }
}

// ================= attention with online softmax (head dim 10) =================
__global__ void k_attention(const float* __restrict__ q, const float* __restrict__ k,
                            const float* __restrict__ v, _Float16* __restrict__ oh)
{
    size_t idx = (size_t)blockIdx.x * blockDim.x + threadIdx.x;
    size_t total = (size_t)BSZ * NHEADc * SEQc;
    if (idx >= total) return;
    int s = (int)(idx % SEQc);
    int h = (int)((idx / SEQc) % NHEADc);
    int b = (int)(idx / ((size_t)SEQc * NHEADc));
    const float* qp = q + ((size_t)b * SEQc + s) * NZc + h * DHEADc;
    float qr[DHEADc];
#pragma unroll
    for (int d = 0; d < DHEADc; ++d) qr[d] = qp[d];
    float m = -1e30f, ssum = 0.f, acc[DHEADc] = {0};
    for (int t = 0; t < SEQc; ++t) {
        const float* kp = k + ((size_t)b * SEQc + t) * NZc + h * DHEADc;
        float sc = 0.f;
#pragma unroll
        for (int d = 0; d < DHEADc; ++d) sc += qr[d] * kp[d];
        sc *= QK_SCALE;
        if (sc > m) {
            float cor = __expf(m - sc);
            ssum *= cor;
#pragma unroll
            for (int d = 0; d < DHEADc; ++d) acc[d] *= cor;
            m = sc;
        }
        float p = __expf(sc - m);
        ssum += p;
        const float* vp = v + ((size_t)b * SEQc + t) * NZc + h * DHEADc;
#pragma unroll
        for (int d = 0; d < DHEADc; ++d) acc[d] += p * vp[d];
    }
    float inv = 1.f / ssum;
    _Float16* op = oh + ((size_t)b * SEQc + s) * 64 + h * DHEADc;
#pragma unroll
    for (int d = 0; d < DHEADc; ++d) op[d] = (_Float16)(acc[d] * inv);
}

// ================= residual add + layernorm (50 dims/row) =================
__global__ void k_ln_res(float* __restrict__ x, const float* __restrict__ t,
                         const float* __restrict__ g, const float* __restrict__ bta,
                         _Float16* __restrict__ xh)
{
    size_t row = (size_t)blockIdx.x * blockDim.x + threadIdx.x;
    if (row >= (size_t)MROWS) return;
    float r[NZc];
    float mean = 0.f;
#pragma unroll
    for (int c = 0; c < NZc; ++c) { r[c] = x[row * NZc + c] + t[row * NZc + c]; mean += r[c]; }
    mean *= (1.f / NZc);
    float var = 0.f;
#pragma unroll
    for (int c = 0; c < NZc; ++c) { float d = r[c] - mean; var += d * d; }
    var *= (1.f / NZc);
    float inv = rsqrtf(var + 1e-5f);
#pragma unroll
    for (int c = 0; c < NZc; ++c) {
        float o = (r[c] - mean) * inv * g[c] + bta[c];
        x[row * NZc + c] = o;
        xh[row * 64 + c] = (_Float16)o;
    }
}

// ================= arcface =================
__global__ void k_arcface(const float* __restrict__ x, const float* __restrict__ W,
                          float* __restrict__ z)
{
    size_t idx = (size_t)blockIdx.x * blockDim.x + threadIdx.x;
    size_t total = 2ull * BSZ * NZc;
    if (idx >= total) return;
    int c = (int)(idx % NZc);
    int b = (int)((idx / NZc) % BSZ);
    int pair = (int)(idx / ((size_t)NZc * BSZ));
    const float* xr = x + ((size_t)b * SEQc + (pair ? (NTOKc + 1) : 0)) * NZc;
    float nx = 0.f;
#pragma unroll
    for (int i = 0; i < NZc; ++i) nx += xr[i] * xr[i];
    nx = sqrtf(nx) + 1e-12f;
    const float* wr = W + (size_t)c * NZc;
    float nw = 0.f, dot = 0.f;
#pragma unroll
    for (int i = 0; i < NZc; ++i) { nw += wr[i] * wr[i]; dot += xr[i] * wr[i]; }
    nw = sqrtf(nw) + 1e-12f;
    float cosv = dot / (nx * nw);
    float s2 = 1.f - cosv * cosv;
    s2 = s2 < 0.f ? 0.f : (s2 > 1.f ? 1.f : s2);
    float sinv = sqrtf(s2);
    float phi = cosv * COS_Mc - sinv * SIN_Mc;
    if (!(cosv > THc)) phi = cosv - MMc;
    float o = ARC_Sc * ((c == pair) ? phi : cosv);
    z[idx] = o;
}

// ================= decoder first layer: z(b,50,1) -> (b,256,128), relu =================
__global__ void k_dec0(const float* __restrict__ z, const float* __restrict__ w,
                       float* __restrict__ out)
{
    size_t idx = (size_t)blockIdx.x * blockDim.x + threadIdx.x;
    size_t total = (size_t)BSZ * 256 * 128;
    if (idx >= total) return;
    int l  = (int)(idx % 128);
    int co = (int)((idx / 128) % 256);
    int b  = (int)(idx / (128 * 256));
    const float* zp = z + (size_t)b * NZc;
    float acc = 0.f;
    for (int ci = 0; ci < NZc; ++ci)
        acc += zp[ci] * w[((size_t)ci * 256 + co) * 128 + l];
    out[idx] = acc > 0.f ? acc : 0.f;
}

// ================= transposed conv (k=4, s=2, p=1), relu =================
__global__ void k_convt(const float* __restrict__ in, const float* __restrict__ w,
                        float* __restrict__ out, int Cin, int Cout, int Lin)
{
    int Lout = Lin * 2;
    size_t idx = (size_t)blockIdx.x * blockDim.x + threadIdx.x;
    size_t total = (size_t)BSZ * Cout * Lout;
    if (idx >= total) return;
    int l  = (int)(idx % Lout);
    int co = (int)((idx / Lout) % Cout);
    int b  = (int)(idx / ((size_t)Lout * Cout));
    float acc = 0.f;
    for (int ci = 0; ci < Cin; ++ci) {
        const float* ip = in + ((size_t)b * Cin + ci) * Lin;
        const float* wp = w + ((size_t)ci * Cout + co) * 4;
#pragma unroll
        for (int j = 0; j < 4; ++j) {
            int t = l + 1 - j;                       // out[l] <- in[(l+1-j)/2] * w[j]
            if (t >= 0 && !(t & 1)) {
                int i = t >> 1;
                if (i < Lin) acc += ip[i] * wp[j];
            }
        }
    }
    out[idx] = acc > 0.f ? acc : 0.f;
}

// ================= decoder last layer: 16ch -> 1ch, tanh, average passes =================
__global__ void k_dec_last(const float* __restrict__ in, const float* __restrict__ w,
                           float* __restrict__ gen1, float* __restrict__ out, int pass)
{
    size_t idx = (size_t)blockIdx.x * blockDim.x + threadIdx.x;
    size_t total = (size_t)BSZ * LSIG;
    if (idx >= total) return;
    int l = (int)(idx % LSIG);
    int b = (int)(idx / LSIG);
    float acc = 0.f;
    for (int ci = 0; ci < 16; ++ci) {
        const float* ip = in + ((size_t)b * 16 + ci) * 4096;
        const float* wp = w + (size_t)ci * 4;
#pragma unroll
        for (int j = 0; j < 4; ++j) {
            int t = l + 1 - j;
            if (t >= 0 && !(t & 1)) {
                int i = t >> 1;
                if (i < 4096) acc += ip[i] * wp[j];
            }
        }
    }
    float v = tanhf(acc);
    if (pass == 0) gen1[idx] = v;
    else           out[idx] = 0.5f * (v + gen1[idx]);
}

// ==========================================================================
extern "C" void kernel_launch(void* const* d_in, const int* in_sizes, int n_in,
                              void* d_out, int out_size, void* d_ws, size_t ws_size,
                              hipStream_t stream)
{
    (void)in_sizes; (void)n_in; (void)out_size; (void)ws_size;
    char* ws = (char*)d_ws;
    float* dout = (float*)d_out;

    float*     bufA = (float*)(ws + OFF_A);
    float*     bufB = (float*)(ws + OFF_B);
    float*     x    = (float*)(ws + OFF_X);
    _Float16*  xh   = (_Float16*)(ws + OFF_XH);
    _Float16*  oh   = (_Float16*)(ws + OFF_OH);
    float*     tmp  = (float*)(ws + OFF_TMP);
    _Float16*  WHb  = (_Float16*)(ws + OFF_WH);
    float*     zbuf = (float*)(ws + OFF_Z);
    float*     gen1 = (float*)(ws + OFF_G1);
    float*     qb   = (float*)(ws + OFF_B);                // q,k,v overlay bufB
    float*     kb   = qb + (size_t)MROWS * NZc;
    float*     vb   = kb + (size_t)MROWS * NZc;
    _Float16*  hid  = (_Float16*)(ws + OFF_A);             // FF hidden overlays bufA

    float* latS = dout + (size_t)BSZ * LSIG;               // 2,097,152
    float* latF = latS + (size_t)BSZ * NZc * NTOKc;        // +1,638,400

    // ---- input indices (setup_inputs flattening order) ----
    const int IN_XSIG = 0, IN_XFREQ = 1, IN_ENCSIG = 2, IN_ENCFRQ = 8;
    const int IN_CLSS = 14, IN_CLSF = 15, IN_POS = 16, IN_TF = 17;
    const int IN_ARCW = 65, IN_DEC = 66;

    // ---- 1. convert transformer weights to padded TRANSPOSED f16 ----
    for (int lt = 0; lt < 3; ++lt) {
        int base = IN_TF + lt * 16;
        _Float16* wd = WHb + (size_t)lt * WH_PER_LAYER;
        k_pad_f16t<<<GRD(64 * 64),  256, 0, stream>>>((const float*)d_in[base + 0], wd,          50, 50, 64, 64);
        k_pad_f16t<<<GRD(64 * 64),  256, 0, stream>>>((const float*)d_in[base + 2], wd + 4096,   50, 50, 64, 64);
        k_pad_f16t<<<GRD(64 * 64),  256, 0, stream>>>((const float*)d_in[base + 4], wd + 8192,   50, 50, 64, 64);
        k_pad_f16t<<<GRD(64 * 64),  256, 0, stream>>>((const float*)d_in[base + 6], wd + 12288,  50, 50, 64, 64);
        k_pad_f16t<<<GRD(64 * 512), 256, 0, stream>>>((const float*)d_in[base + 8], wd + 16384,  50, 512, 64, 512);
        k_pad_f16t<<<GRD(512 * 64), 256, 0, stream>>>((const float*)d_in[base +10], wd + 49152, 512, 50, 512, 64);
    }

    // ---- 2. encoders ----
    const int encC[7] = {1, 16, 32, 64, 64, 64, 50};
    const int encL[7] = {8192, 4096, 2048, 1024, 512, 256, 128};
    for (int pass = 0; pass < 2; ++pass) {
        const float* cur = (const float*)d_in[pass ? IN_XFREQ : IN_XSIG];
        int wbase = pass ? IN_ENCFRQ : IN_ENCSIG;
        for (int i = 0; i < 6; ++i) {
            float* outp = (i == 5) ? (pass ? latF : latS) : ((i & 1) ? bufB : bufA);
            size_t n = (size_t)BSZ * encC[i + 1] * encL[i + 1];
            k_conv1d<<<GRD(n), 256, 0, stream>>>(cur, (const float*)d_in[wbase + i],
                                                 outp, encC[i], encC[i + 1], encL[i], (i < 5) ? 1 : 0);
            cur = outp;
        }
    }

    // ---- 3. build tokens ----
    k_build_x<<<GRD((size_t)MROWS * 64), 256, 0, stream>>>(
        latS, latF, (const float*)d_in[IN_CLSS], (const float*)d_in[IN_CLSF],
        (const float*)d_in[IN_POS], x, xh);
    k_zero_h<<<GRD((size_t)MROWS * 64), 256, 0, stream>>>(oh, (size_t)MROWS * 64);

    auto gemm = [&](const _Float16* A, int lda, const _Float16* Bt, int ldbt,
                    const float* bias, float* oF, int ldoF, _Float16* oHp, int ldoH,
                    int N, int K, int nValid, int relu) {
        int groups = (MROWS / 16) * (N / 64);
        k_gemm_wmma<<<(groups + 3) / 4, 128, 0, stream>>>(A, lda, Bt, ldbt, bias,
                                                          oF, ldoF, oHp, ldoH,
                                                          MROWS, N, K, nValid, relu);
    };

    // ---- 4. transformer layers ----
    for (int lt = 0; lt < 3; ++lt) {
        int base = IN_TF + lt * 16;
        _Float16* wd = WHb + (size_t)lt * WH_PER_LAYER;
        const float* bq = (const float*)d_in[base + 1];
        const float* bk = (const float*)d_in[base + 3];
        const float* bv = (const float*)d_in[base + 5];
        const float* bo = (const float*)d_in[base + 7];
        const float* b1 = (const float*)d_in[base + 9];
        const float* b2 = (const float*)d_in[base + 11];

        gemm(xh, 64, wd,         64, bq, qb, NZc, nullptr, 0, 64, 64, NZc, 0);
        gemm(xh, 64, wd + 4096,  64, bk, kb, NZc, nullptr, 0, 64, 64, NZc, 0);
        gemm(xh, 64, wd + 8192,  64, bv, vb, NZc, nullptr, 0, 64, 64, NZc, 0);
        k_attention<<<GRD((size_t)BSZ * NHEADc * SEQc), 256, 0, stream>>>(qb, kb, vb, oh);
        gemm(oh, 64, wd + 12288, 64, bo, tmp, NZc, nullptr, 0, 64, 64, NZc, 0);
        k_ln_res<<<GRD((size_t)MROWS), 256, 0, stream>>>(
            x, tmp, (const float*)d_in[base + 12], (const float*)d_in[base + 13], xh);
        gemm(xh, 64, wd + 16384, 64, b1, nullptr, 0, hid, 512, 512, 64, 512, 1);
        gemm(hid, 512, wd + 49152, 512, b2, tmp, NZc, nullptr, 0, 64, 512, NZc, 0);
        k_ln_res<<<GRD((size_t)MROWS), 256, 0, stream>>>(
            x, tmp, (const float*)d_in[base + 14], (const float*)d_in[base + 15], xh);
    }

    // ---- 5. arcface ----
    k_arcface<<<GRD(2ull * BSZ * NZc), 256, 0, stream>>>(x, (const float*)d_in[IN_ARCW], zbuf);

    // ---- 6. decoder (two passes, averaged) ----
    const int mdCi[5] = {256, 128, 64, 64, 32};
    const int mdCo[5] = {128, 64, 64, 32, 16};
    const int mdLi[5] = {128, 256, 512, 1024, 2048};
    for (int pass = 0; pass < 2; ++pass) {
        k_dec0<<<GRD((size_t)BSZ * 256 * 128), 256, 0, stream>>>(
            zbuf + (size_t)pass * BSZ * NZc, (const float*)d_in[IN_DEC], bufA);
        float* pin = bufA;
        float* pout = bufB;
        for (int i = 0; i < 5; ++i) {
            size_t n = (size_t)BSZ * mdCo[i] * mdLi[i] * 2;
            k_convt<<<GRD(n), 256, 0, stream>>>(pin, (const float*)d_in[IN_DEC + 1 + i],
                                                pout, mdCi[i], mdCo[i], mdLi[i]);
            float* t = pin; pin = pout; pout = t;
        }
        k_dec_last<<<GRD((size_t)BSZ * LSIG), 256, 0, stream>>>(
            pin, (const float*)d_in[IN_DEC + 6], gen1, dout, pass);
    }
}
